// RoCo_28999619183213
// MI455X (gfx1250) — matrix operand order repaired and checked by reading, not compiled, not run on hardware
//
#include <hip/hip_runtime.h>
#include <hip/hip_bf16.h>

// ---------------------------------------------------------------------------
// Causal flash attention + RoCo column statistics for MI455X (gfx1250).
// B=1, H=32, Q=KV=2048, D=128, fp32 in/out.  Compute path: bf16 WMMA
// (v_wmma_f32_16x16x32_bf16) with f32 accumulation; exp2-domain online
// softmax in f32 VALU with DPP16 butterfly reductions.  Independent WMMA
// chains are interleaved so LDS fragment-load latency overlaps the XDL pipe.
// ---------------------------------------------------------------------------

typedef __attribute__((ext_vector_type(16))) __bf16 v16bf;
typedef __attribute__((ext_vector_type(8)))  float  v8f;

union BF16Frag {
  v16bf  v;
  __bf16 b[16];
  uint4  u[2];
};

#define HEADS 32
#define DIM   128
#define SEQ   2048
#define KSTR  136   // bf16 elems per K-tile row (128 + 8 pad)
#define VSTR  40    // bf16 elems per V^T-tile row (32 + 8 pad)
#define PSTR  40    // bf16 elems per P-stage row   (32 + 8 pad)
#define NEGF  (-1e30f)

// DPP16 controls
#define DPP_QPERM_XOR1  0xB1   // quad_perm [1,0,3,2]
#define DPP_QPERM_XOR2  0x4E   // quad_perm [2,3,0,1]
#define DPP_ROW_HMIRROR 0x141  // mirror within 8-lane half-row
#define DPP_ROW_MIRROR  0x140  // mirror within 16-lane row

template <int CTRL>
__device__ __forceinline__ float dpp_f32(float x) {
  int xi = __builtin_bit_cast(int, x);
  int r  = __builtin_amdgcn_update_dpp(0, xi, CTRL, 0xF, 0xF, true);
  return __builtin_bit_cast(float, r);
}

// Reduce (max / sum) across the 16 lanes of each half-row; every lane of the
// half ends up holding the reduced value.
__device__ __forceinline__ float red_max16(float x) {
  x = fmaxf(x, dpp_f32<DPP_QPERM_XOR1>(x));
  x = fmaxf(x, dpp_f32<DPP_QPERM_XOR2>(x));
  x = fmaxf(x, dpp_f32<DPP_ROW_HMIRROR>(x));
  x = fmaxf(x, dpp_f32<DPP_ROW_MIRROR>(x));
  return x;
}
__device__ __forceinline__ float red_sum16(float x) {
  x += dpp_f32<DPP_QPERM_XOR1>(x);
  x += dpp_f32<DPP_QPERM_XOR2>(x);
  x += dpp_f32<DPP_ROW_HMIRROR>(x);
  x += dpp_f32<DPP_ROW_MIRROR>(x);
  return x;
}

__device__ __forceinline__ v8f wmma_bf16(v16bf a, v16bf b, v8f c) {
  // (neg_a, A, neg_b, B, c_mod, C, reuse_a, reuse_b)
  return __builtin_amdgcn_wmma_f32_16x16x32_bf16(false, a, false, b,
                                                 (short)0, c, false, false);
}

__device__ __forceinline__ void load_frag(BF16Frag& f, const __bf16* p, int gap) {
  f.u[0] = *(const uint4*)p;
  f.u[1] = *(const uint4*)(p + gap);
}

__device__ __forceinline__ void pack8(BF16Frag& f, int base, float4 a, float4 b) {
  f.b[base + 0] = (__bf16)a.x; f.b[base + 1] = (__bf16)a.y;
  f.b[base + 2] = (__bf16)a.z; f.b[base + 3] = (__bf16)a.w;
  f.b[base + 4] = (__bf16)b.x; f.b[base + 5] = (__bf16)b.y;
  f.b[base + 6] = (__bf16)b.z; f.b[base + 7] = (__bf16)b.w;
}

__global__ void roco_zero_stats(float* p, int n) {
  int i = blockIdx.x * blockDim.x + threadIdx.x;
  if (i < n) p[i] = 0.0f;
}

__global__ __launch_bounds__(256)
void roco_attn_fwd(const float* __restrict__ qg, const float* __restrict__ kg,
                   const float* __restrict__ vg, float* __restrict__ outg,
                   float* __restrict__ score, float* __restrict__ scoresq) {
  __shared__ __align__(16) __bf16 Klds[32 * KSTR];     // K tile, row-major [kv][d]
  __shared__ __align__(16) __bf16 Vlds[DIM * VSTR];    // V tile transposed [d][kv]
  __shared__ __align__(16) __bf16 Plds[8 * 16 * PSTR]; // per-wave P stage [row][kv]

  const int  tid   = threadIdx.x;
  const int  wave  = tid >> 5;
  const int  lane  = tid & 31;
  const int  lo    = lane & 15;
  const bool hi    = lane >= 16;
  const int  khi   = hi ? 16 : 0;
  const int  h     = blockIdx.y;
  const int  qbase = blockIdx.x * 128;
  const int  q0    = qbase + wave * 16;
  // exp2-domain scale: (1/sqrt(128)) * log2(e)
  const float scale2 = 0.088388347648318447f * 1.4426950408889634f;

  __bf16* Pw = &Plds[wave * 16 * PSTR];

  // ---- preload Q tile as 4 A-matrix fragments (D chunks of 32), bf16 ----
  BF16Frag aq[4];
  {
    const float* qrow = qg + ((size_t)(q0 + lo) * HEADS + h) * DIM;
#pragma unroll
    for (int c = 0; c < 4; ++c) {
      int d1 = c * 32 + (hi ? 16 : 0);   // K elems 0..7 of fragment
      int d2 = c * 32 + (hi ? 24 : 8);   // K elems 8..15 of fragment
      float4 f0 = *(const float4*)(qrow + d1);
      float4 f1 = *(const float4*)(qrow + d1 + 4);
      float4 f2 = *(const float4*)(qrow + d2);
      float4 f3 = *(const float4*)(qrow + d2 + 4);
      pack8(aq[c], 0, f0, f1);
      pack8(aq[c], 8, f2, f3);
    }
  }

  // ---- accumulators ----
  v8f ofrag[8];
#pragma unroll
  for (int g = 0; g < 8; ++g)
#pragma unroll
    for (int r = 0; r < 8; ++r) ofrag[g][r] = 0.0f;
  float m_run[8], l_run[8];
#pragma unroll
  for (int r = 0; r < 8; ++r) { m_run[r] = NEGF; l_run[r] = 0.0f; }

  const int nsteps = (qbase >> 5) + 4; // covers kv in [0, qbase+128)

  // Per-lane LDS fragment base pointers (constant across steps).
  const __bf16* kbase0 = &Klds[lo * KSTR + khi];          // nh = 0 columns
  const __bf16* kbase1 = &Klds[(16 + lo) * KSTR + khi];   // nh = 1 columns

  // =======================  PASS 1: O, m, l  =======================
  for (int s = 0; s < nsteps; ++s) {
    const int kv0 = s * 32;
    __syncthreads(); // previous-iteration LDS reads done

    // Cooperative staging: K[32][128] and V^T[128][32] as bf16.
#pragma unroll
    for (int i = 0; i < 4; ++i) {
      int fi  = tid + i * 256;       // float4 index, 0..1023
      int row = fi >> 5;             // kv_local
      int c4  = fi & 31;
      int d   = c4 * 4;
      size_t goff = ((size_t)(kv0 + row) * HEADS + h) * DIM + d;
      float4 kf = *(const float4*)(kg + goff);
      float4 vf = *(const float4*)(vg + goff);
      union { __bf16 b[4]; uint2 u; } pk;
      pk.b[0] = (__bf16)kf.x; pk.b[1] = (__bf16)kf.y;
      pk.b[2] = (__bf16)kf.z; pk.b[3] = (__bf16)kf.w;
      *(uint2*)&Klds[row * KSTR + d] = pk.u;
      float fv[4] = {vf.x, vf.y, vf.z, vf.w};
#pragma unroll
      for (int j = 0; j < 4; ++j)
        Vlds[(d + j) * VSTR + row] = (__bf16)fv[j];
      if (s + 1 < nsteps)
        __builtin_prefetch(kg + goff + (size_t)32 * HEADS * DIM, 0, 1);
    }
    __syncthreads();

    if (kv0 <= q0 + 15) { // wave-uniform causal activity test
      // ---- S = Q * K^T : two interleaved independent WMMA chains ----
      v8f acc0, acc1;
#pragma unroll
      for (int r = 0; r < 8; ++r) { acc0[r] = 0.0f; acc1[r] = 0.0f; }
#pragma unroll
      for (int c = 0; c < 4; ++c) {
        BF16Frag b0, b1;
        load_frag(b0, kbase0 + c * 32, 8);
        load_frag(b1, kbase1 + c * 32, 8);
        acc0 = wmma_bf16(aq[c].v, b0.v, acc0);
        acc1 = wmma_bf16(aq[c].v, b1.v, acc1);
      }

      // ---- exp2-domain online softmax (DPP reductions); stage P ----
      const int kvA = kv0 + lo;
      const int kvB = kvA + 16;
#pragma unroll
      for (int r = 0; r < 8; ++r) {
        int   qrow = q0 + r + (hi ? 8 : 0);
        float s0 = (kvA > qrow) ? NEGF : acc0[r] * scale2;
        float s1 = (kvB > qrow) ? NEGF : acc1[r] * scale2;
        float rm    = red_max16(fmaxf(s0, s1));
        float mnew  = fmaxf(m_run[r], rm);
        float alpha = __builtin_amdgcn_exp2f(m_run[r] - mnew);
        float p0 = __builtin_amdgcn_exp2f(s0 - mnew);
        float p1 = __builtin_amdgcn_exp2f(s1 - mnew);
        float rs = red_sum16(p0 + p1);
        l_run[r] = l_run[r] * alpha + rs;
        m_run[r] = mnew;
#pragma unroll
        for (int g = 0; g < 8; ++g) ofrag[g][r] *= alpha;
        int prow = r + (hi ? 8 : 0);
        Pw[prow * PSTR + lo]      = (__bf16)p0;
        Pw[prow * PSTR + 16 + lo] = (__bf16)p1;
      }
      asm volatile("s_wait_dscnt 0x0" ::: "memory");

      // ---- O += P * V : pairs of independent WMMAs ----
      BF16Frag ap;
      load_frag(ap, &Pw[lo * PSTR + khi], 8);
#pragma unroll
      for (int g = 0; g < 8; g += 2) {
        BF16Frag bv0, bv1;
        load_frag(bv0, &Vlds[(g * 16 + lo) * VSTR + khi], 8);
        load_frag(bv1, &Vlds[((g + 1) * 16 + lo) * VSTR + khi], 8);
        ofrag[g]     = wmma_bf16(ap.v, bv0.v, ofrag[g]);
        ofrag[g + 1] = wmma_bf16(ap.v, bv1.v, ofrag[g + 1]);
      }
    }
  }

  // ---- finalize and write O ----
  float linv[8];
#pragma unroll
  for (int r = 0; r < 8; ++r) linv[r] = 1.0f / l_run[r];
#pragma unroll
  for (int r = 0; r < 8; ++r) {
    int row = q0 + r + (hi ? 8 : 0);
    float* op = outg + ((size_t)row * HEADS + h) * DIM + lo;
#pragma unroll
    for (int g = 0; g < 8; ++g) op[g * 16] = ofrag[g][r] * linv[r];
  }

  // ==========  PASS 2: recompute S, accumulate column stats  ==========
  for (int s = 0; s < nsteps; ++s) {
    const int kv0 = s * 32;
    __syncthreads();
#pragma unroll
    for (int i = 0; i < 4; ++i) {
      int fi  = tid + i * 256;
      int row = fi >> 5;
      int c4  = fi & 31;
      int d   = c4 * 4;
      size_t goff = ((size_t)(kv0 + row) * HEADS + h) * DIM + d;
      float4 kf = *(const float4*)(kg + goff);
      union { __bf16 b[4]; uint2 u; } pk;
      pk.b[0] = (__bf16)kf.x; pk.b[1] = (__bf16)kf.y;
      pk.b[2] = (__bf16)kf.z; pk.b[3] = (__bf16)kf.w;
      *(uint2*)&Klds[row * KSTR + d] = pk.u;
    }
    __syncthreads();

    if (kv0 <= q0 + 15) {
      v8f acc0, acc1;
#pragma unroll
      for (int r = 0; r < 8; ++r) { acc0[r] = 0.0f; acc1[r] = 0.0f; }
#pragma unroll
      for (int c = 0; c < 4; ++c) {
        BF16Frag b0, b1;
        load_frag(b0, kbase0 + c * 32, 8);
        load_frag(b1, kbase1 + c * 32, 8);
        acc0 = wmma_bf16(aq[c].v, b0.v, acc0);
        acc1 = wmma_bf16(aq[c].v, b1.v, acc1);
      }
      const int kvA = kv0 + lo;
      const int kvB = kvA + 16;
      float cs0 = 0.0f, cq0 = 0.0f, cs1 = 0.0f, cq1 = 0.0f;
#pragma unroll
      for (int r = 0; r < 8; ++r) {
        int   qrow = q0 + r + (hi ? 8 : 0);
        float p0 = (kvA > qrow)
                       ? 0.0f
                       : __builtin_amdgcn_exp2f(acc0[r] * scale2 - m_run[r]) *
                             linv[r];
        float p1 = (kvB > qrow)
                       ? 0.0f
                       : __builtin_amdgcn_exp2f(acc1[r] * scale2 - m_run[r]) *
                             linv[r];
        cs0 += p0; cq0 += p0 * p0;
        cs1 += p1; cq1 += p1 * p1;
      }
      atomicAdd(score   + (size_t)h * SEQ + kvA, cs0);
      atomicAdd(scoresq + (size_t)h * SEQ + kvA, cq0);
      atomicAdd(score   + (size_t)h * SEQ + kvB, cs1);
      atomicAdd(scoresq + (size_t)h * SEQ + kvB, cq1);
    }
  }
}

extern "C" void kernel_launch(void* const* d_in, const int* in_sizes, int n_in,
                              void* d_out, int out_size, void* d_ws, size_t ws_size,
                              hipStream_t stream) {
  (void)in_sizes; (void)n_in; (void)out_size; (void)d_ws; (void)ws_size;
  const float* q = (const float*)d_in[0];
  const float* k = (const float*)d_in[1];
  const float* v = (const float*)d_in[2];
  float* out     = (float*)d_out;
  float* score   = out + (size_t)SEQ * HEADS * DIM; // [H, KV]
  float* scoresq = score + (size_t)HEADS * SEQ;     // [H, KV]

  // Stats are accumulated with atomics -> zero them first (d_out is poisoned).
  int nstat = 2 * HEADS * SEQ;
  roco_zero_stats<<<(nstat + 255) / 256, 256, 0, stream>>>(score, nstat);

  dim3 grid(SEQ / 128, HEADS);
  roco_attn_fwd<<<grid, 256, 0, stream>>>(q, k, v, out, score, scoresq);
}
